// linear_projector_42142219108609
// MI455X (gfx1250) — matrix operand order repaired and checked by reading, not compiled, not run on hardware
//
#include <hip/hip_runtime.h>
#include <hip/hip_bf16.h>

typedef __attribute__((ext_vector_type(16))) __bf16 v16bf;
typedef __attribute__((ext_vector_type(8)))  float  v8f;

struct U4 { unsigned int x, y, z, w; };
union BFrag { v16bf v; U4 q[2]; };

__device__ __forceinline__ unsigned short f2bf(float f) {
  unsigned int u = __float_as_uint(f);
  u += 0x7FFFu + ((u >> 16) & 1u);   // round-to-nearest-even
  return (unsigned short)(u >> 16);
}

__device__ __forceinline__ v8f wmma_bf16(const BFrag& a, const BFrag& b, v8f c) {
  return __builtin_amdgcn_wmma_f32_16x16x32_bf16(false, a.v, false, b.v,
                                                 (short)0, c, false, false);
}

// ---------------------------------------------------------------- converts
__global__ __launch_bounds__(256)
void k_f32_to_bf16(const float* __restrict__ src, unsigned short* __restrict__ dst, int n) {
  int i = blockIdx.x * 256 + threadIdx.x;
  int stride = gridDim.x * 256;
  for (; i < n; i += stride) dst[i] = f2bf(src[i]);
}

// ------------------------------------------------------- param fusion (BN)
__global__ __launch_bounds__(256)
void k_prep(const float* __restrict__ conv_b, const float* __restrict__ bn_g,
            const float* __restrict__ bn_b, const float* __restrict__ bn_m,
            const float* __restrict__ bn_v,
            const float* __restrict__ lin_w_img, const float* __restrict__ lin_b_img,
            const float* __restrict__ lin_w_pc,  const float* __restrict__ lin_b_pc,
            const float* __restrict__ pe_b1, const float* __restrict__ pe_g1,
            const float* __restrict__ pe_bb1, const float* __restrict__ pe_m1,
            const float* __restrict__ pe_v1,
            const float* __restrict__ pe_w2, const float* __restrict__ pe_b2,
            const float* __restrict__ pe_g2, const float* __restrict__ pe_bb2,
            const float* __restrict__ pe_m2, const float* __restrict__ pe_v2,
            float* __restrict__ scale, float* __restrict__ beta,
            float* __restrict__ linb, unsigned short* __restrict__ lwbf,
            float* __restrict__ pe_s1, float* __restrict__ pe_t1,
            float* __restrict__ pe_s2, float* __restrict__ pe_t2,
            unsigned short* __restrict__ w2bf) {
  const float EPS = 1e-5f;
  int tid = blockIdx.x * 256 + threadIdx.x;
  int stride = gridDim.x * 256;
  for (int i = tid; i < 4 * 512; i += stride) {
    float s = bn_g[i] * rsqrtf(bn_v[i] + EPS);
    scale[i] = s;
    beta[i] = (conv_b[i] - bn_m[i]) * s + bn_b[i];
  }
  for (int i = tid; i < 128; i += stride) {
    int mod = i >> 5, n = i & 31;
    linb[i] = (mod < 2) ? lin_b_img[mod * 32 + n] : lin_b_pc[(mod - 2) * 32 + n];
  }
  for (int i = tid; i < 4 * 32 * 64; i += stride) {   // zero-padded [4,32,64]
    int k = i & 63, n = (i >> 6) & 31, mod = i >> 11;
    float v = 0.f;
    if (mod < 2) { if (k < 49) v = lin_w_img[(mod * 32 + n) * 49 + k]; }
    else         { if (k < 32) v = lin_w_pc[((mod - 2) * 32 + n) * 32 + k]; }
    lwbf[i] = f2bf(v);
  }
  for (int i = tid; i < 128; i += stride) {
    float s = pe_g1[i] * rsqrtf(pe_v1[i] + EPS);
    pe_s1[i] = s;
    pe_t1[i] = (pe_b1[i] - pe_m1[i]) * s + pe_bb1[i];
  }
  for (int i = tid; i < 512; i += stride) {
    float s = pe_g2[i] * rsqrtf(pe_v2[i] + EPS);
    pe_s2[i] = s;
    pe_t2[i] = (pe_b2[i] - pe_m2[i]) * s + pe_bb2[i];
  }
  for (int i = tid; i < 512 * 128; i += stride) w2bf[i] = f2bf(pe_w2[i]);
}

// -------------------- branch GEMM: h[b,o,l] = relu(scale*(x@W^T) + beta) ---
// Each wave computes a 16(M) x 64(N) register tile (4 accumulators): the A
// fragment is loaded once per K-step and reused across 4 WMMAs. One block
// (8 waves) covers all 512 output channels for one 16-row M tile.
__global__ __launch_bounds__(256)
void k_branch_gemm(const unsigned short* __restrict__ xbf,
                   const unsigned short* __restrict__ wbf,
                   const float* __restrict__ scale, const float* __restrict__ beta,
                   unsigned short* __restrict__ h) {
  int mod = blockIdx.y;
  int L = (mod < 2) ? 49 : 32;
  int Mrows = 256 * L;
  int mbase = blockIdx.x * 16;
  if (mbase >= Mrows) return;
  int lane = threadIdx.x & 31, wave = threadIdx.x >> 5;
  int hi = lane >> 4, l15 = lane & 15;
  int nbase = wave * 64;   // 8 waves * 64 cols = 512

  size_t xoff = (size_t)((mod < 2) ? mod * 12544 : 25088 + (mod - 2) * 8192) * 512;
  const unsigned short* xrow = xbf + xoff + (size_t)(mbase + l15) * 512;
  const unsigned short* w0 = wbf + (size_t)mod * 512 * 512 + (size_t)(nbase + l15) * 512;

  v8f acc[4] = {};
  for (int kk = 0; kk < 16; ++kk) {
    int k0 = kk * 32;
    BFrag a;
    a.q[0] = *(const U4*)(xrow + k0 + hi * 16);       // K = [hi*16, hi*16+8)
    a.q[1] = *(const U4*)(xrow + k0 + hi * 16 + 8);   // K = [hi*16+8, hi*16+16)
#pragma unroll
    for (int t = 0; t < 4; ++t) {
      const unsigned short* bp = w0 + (size_t)t * 16 * 512 + k0 + hi * 16;
      BFrag b;
      b.q[0] = *(const U4*)(bp);                      // K = [hi*16, hi*16+16)
      b.q[1] = *(const U4*)(bp + 8);
      acc[t] = wmma_bf16(a, b, acc[t]);
    }
  }
  unsigned short* hmod = h + (size_t)mod * 256 * 512 * 64;
  float sc[4], bt[4];
#pragma unroll
  for (int t = 0; t < 4; ++t) {
    int o = nbase + t * 16 + l15;
    sc[t] = scale[mod * 512 + o];
    bt[t] = beta[mod * 512 + o];
  }
#pragma unroll
  for (int v = 0; v < 8; ++v) {
    int m = mbase + v + hi * 8;          // D: row M = v + hi*8 ; col N = lane&15
    int bidx = m / L;
    int li = m - bidx * L;
    size_t rowbase = (size_t)bidx * 512 * 64 + li;
#pragma unroll
    for (int t = 0; t < 4; ++t) {
      int o = nbase + t * 16 + l15;
      float val = fmaxf(acc[t][v] * sc[t] + bt[t], 0.f);
      hmod[rowbase + (size_t)o * 64] = f2bf(val);
    }
  }
}

// ---------------- token Linear: out[b,tok,o] = relu(h[b,o,:] @ lw^T + lb) --
// Each wave computes 16(M=o) x 32(N=both token tiles), reusing A per K-step.
__global__ __launch_bounds__(256)
void k_token_linear(const unsigned short* __restrict__ h,
                    const unsigned short* __restrict__ lwbf,
                    const float* __restrict__ linb, float* __restrict__ out) {
  int mod = blockIdx.z, b = blockIdx.y;
  int lane = threadIdx.x & 31, wave = threadIdx.x >> 5;
  int hi = lane >> 4, l15 = lane & 15;
  int mbase = blockIdx.x * 128 + wave * 16;   // output channel o (M)

  const unsigned short* arow =
      h + (size_t)mod * 256 * 512 * 64 + ((size_t)b * 512 + (mbase + l15)) * 64;
  const unsigned short* brow0 = lwbf + (size_t)(mod * 32 + l15) * 64;        // tokens 0..15
  const unsigned short* brow1 = lwbf + (size_t)(mod * 32 + 16 + l15) * 64;   // tokens 16..31

  v8f acc0 = {}, acc1 = {};
#pragma unroll
  for (int kk = 0; kk < 2; ++kk) {
    int k0 = kk * 32;
    BFrag a, b0, b1;
    a.q[0] = *(const U4*)(arow + k0 + hi * 16);
    a.q[1] = *(const U4*)(arow + k0 + hi * 16 + 8);
    b0.q[0] = *(const U4*)(brow0 + k0 + hi * 16);
    b0.q[1] = *(const U4*)(brow0 + k0 + hi * 16 + 8);
    b1.q[0] = *(const U4*)(brow1 + k0 + hi * 16);
    b1.q[1] = *(const U4*)(brow1 + k0 + hi * 16 + 8);
    acc0 = wmma_bf16(a, b0, acc0);
    acc1 = wmma_bf16(a, b1, acc1);
  }
  float lb0 = linb[mod * 32 + l15];
  float lb1 = linb[mod * 32 + 16 + l15];
  int tok0 = mod * 32 + l15;
  int tok1 = mod * 32 + 16 + l15;
#pragma unroll
  for (int v = 0; v < 8; ++v) {
    int o = mbase + v + hi * 8;
    out[((size_t)b * 128 + tok0) * 512 + o] = fmaxf(acc0[v] + lb0, 0.f);
    out[((size_t)b * 128 + tok1) * 512 + o] = fmaxf(acc1[v] + lb1, 0.f);
  }
}

// --------------------- FPS: one WGP per batch, LDS-resident point cloud ---
__global__ __launch_bounds__(256)
void k_fps(const float* __restrict__ pts, float* __restrict__ nxyz) {
  __shared__ float sx[4096], sy[4096], sz[4096], dist[4096];
  __shared__ float rv[256];
  __shared__ int   ri[256];
  __shared__ int   sfar;
  int b = blockIdx.x, tid = threadIdx.x;
  const float* p = pts + (size_t)b * 4096 * 3;
  for (int i = tid; i < 4096; i += 256) {
    sx[i] = p[i * 3 + 0]; sy[i] = p[i * 3 + 1]; sz[i] = p[i * 3 + 2];
    dist[i] = 1e10f;
  }
  if (tid == 0) sfar = 0;
  __syncthreads();
  for (int it = 0; it < 128; ++it) {
    int far = sfar;
    float cx = sx[far], cy = sy[far], cz = sz[far];
    if (tid == 0) {
      nxyz[((size_t)b * 128 + it) * 3 + 0] = cx;
      nxyz[((size_t)b * 128 + it) * 3 + 1] = cy;
      nxyz[((size_t)b * 128 + it) * 3 + 2] = cz;
    }
    float best = -1.f; int bi = 0;
    for (int i = tid; i < 4096; i += 256) {
      float dx = sx[i] - cx, dy = sy[i] - cy, dz = sz[i] - cz;
      float d = dx * dx + dy * dy + dz * dz;
      float nd = fminf(dist[i], d);
      dist[i] = nd;
      if (nd > best) { best = nd; bi = i; }
    }
    rv[tid] = best; ri[tid] = bi;
    __syncthreads();
    for (int s = 128; s > 0; s >>= 1) {
      if (tid < s && rv[tid + s] > rv[tid]) { rv[tid] = rv[tid + s]; ri[tid] = ri[tid + s]; }
      __syncthreads();
    }
    if (tid == 0) sfar = ri[0];
    __syncthreads();
  }
}

// ----- PE MLP: layer1 (K=3) VALU -> LDS bf16; layer2 WMMA; += into out ----
// Each wave owns one M-tile; all 4 A fragments (K=128) stay register-resident
// across the whole 32-iteration N loop.
__global__ __launch_bounds__(256)
void k_posenc(const float* __restrict__ nxyz, const float* __restrict__ pe_w1,
              const float* __restrict__ s1, const float* __restrict__ t1,
              const unsigned short* __restrict__ w2bf,
              const float* __restrict__ s2, const float* __restrict__ t2,
              float* __restrict__ out) {
  __shared__ float sxyz[128 * 3];
  __shared__ float sw1[128 * 3];
  __shared__ unsigned short h1[128 * 128];
  int b = blockIdx.x, tid = threadIdx.x;
  for (int i = tid; i < 384; i += 256) {
    sxyz[i] = nxyz[(size_t)b * 384 + i];
    sw1[i] = pe_w1[i];
  }
  __syncthreads();
  for (int e = tid; e < 128 * 128; e += 256) {
    int s = e >> 7, c = e & 127;
    float v = sxyz[s * 3 + 0] * sw1[c * 3 + 0] + sxyz[s * 3 + 1] * sw1[c * 3 + 1] +
              sxyz[s * 3 + 2] * sw1[c * 3 + 2];
    v = fmaxf(v * s1[c] + t1[c], 0.f);
    h1[s * 128 + c] = f2bf(v);
  }
  __syncthreads();
  int lane = tid & 31, wave = tid >> 5;
  int hi = lane >> 4, l15 = lane & 15;

  BFrag afr[4];
  const unsigned short* arow = h1 + (size_t)(wave * 16 + l15) * 128;
#pragma unroll
  for (int kk = 0; kk < 4; ++kk) {
    afr[kk].q[0] = *(const U4*)(arow + kk * 32 + hi * 16);
    afr[kk].q[1] = *(const U4*)(arow + kk * 32 + hi * 16 + 8);
  }

  for (int nt = 0; nt < 32; ++nt) {
    const unsigned short* brow = w2bf + (size_t)(nt * 16 + l15) * 128;
    v8f acc = {};
#pragma unroll
    for (int kk = 0; kk < 4; ++kk) {
      BFrag bb;
      bb.q[0] = *(const U4*)(brow + kk * 32 + hi * 16);
      bb.q[1] = *(const U4*)(brow + kk * 32 + hi * 16 + 8);
      acc = wmma_bf16(afr[kk], bb, acc);
    }
    int o = nt * 16 + l15;
    float sc = s2[o], bt = t2[o];
#pragma unroll
    for (int v = 0; v < 8; ++v) {
      int s = wave * 16 + v + hi * 8;
      size_t idx = ((size_t)b * 128 + s) * 512 + o;
      out[idx] += fmaxf(acc[v] * sc + bt, 0.f);
    }
  }
}

// --------------------------------------------------------------- launcher
extern "C" void kernel_launch(void* const* d_in, const int* in_sizes, int n_in,
                              void* d_out, int out_size, void* d_ws, size_t ws_size,
                              hipStream_t stream) {
  const float* rgb        = (const float*)d_in[0];
  const float* depth      = (const float*)d_in[1];
  const float* mmwave     = (const float*)d_in[2];
  const float* lidar      = (const float*)d_in[3];
  const float* lidar_pts  = (const float*)d_in[4];
  const float* conv_w     = (const float*)d_in[5];
  const float* conv_b     = (const float*)d_in[6];
  const float* bn_g       = (const float*)d_in[7];
  const float* bn_b       = (const float*)d_in[8];
  const float* bn_m       = (const float*)d_in[9];
  const float* bn_v       = (const float*)d_in[10];
  const float* lin_w_img  = (const float*)d_in[11];
  const float* lin_b_img  = (const float*)d_in[12];
  const float* lin_w_pc   = (const float*)d_in[13];
  const float* lin_b_pc   = (const float*)d_in[14];
  const float* pe_w1      = (const float*)d_in[15];
  const float* pe_b1      = (const float*)d_in[16];
  const float* pe_g1      = (const float*)d_in[17];
  const float* pe_bb1     = (const float*)d_in[18];
  const float* pe_m1      = (const float*)d_in[19];
  const float* pe_v1      = (const float*)d_in[20];
  const float* pe_w2      = (const float*)d_in[21];
  const float* pe_b2      = (const float*)d_in[22];
  const float* pe_g2      = (const float*)d_in[23];
  const float* pe_bb2     = (const float*)d_in[24];
  const float* pe_m2      = (const float*)d_in[25];
  const float* pe_v2      = (const float*)d_in[26];
  float* out = (float*)d_out;

  char* ws = (char*)d_ws;
  size_t off = 0;
  auto take = [&](size_t bytes) -> char* {
    char* p = ws + off;
    off = (off + bytes + 255) & ~(size_t)255;
    return p;
  };
  const size_t M_IMG = 12544, M_PC = 8192, M_ALL = 2 * M_IMG + 2 * M_PC;  // 41472
  unsigned short* wbf   = (unsigned short*)take(4ull * 512 * 512 * 2);
  unsigned short* xbf   = (unsigned short*)take(M_ALL * 512 * 2);
  unsigned short* hbuf  = (unsigned short*)take(4ull * 256 * 512 * 64 * 2);
  unsigned short* lwbf  = (unsigned short*)take(4ull * 32 * 64 * 2);
  unsigned short* w2bf  = (unsigned short*)take(512ull * 128 * 2);
  float* scale = (float*)take(4ull * 512 * 4);
  float* beta  = (float*)take(4ull * 512 * 4);
  float* linb  = (float*)take(128 * 4);
  float* pe_s1 = (float*)take(128 * 4);
  float* pe_t1 = (float*)take(128 * 4);
  float* pe_s2 = (float*)take(512 * 4);
  float* pe_t2 = (float*)take(512 * 4);
  float* nxyz  = (float*)take(256ull * 128 * 3 * 4);
  (void)ws_size; (void)n_in; (void)in_sizes; (void)out_size;

  // f32 -> bf16 staging
  k_f32_to_bf16<<<1024, 256, 0, stream>>>(conv_w, wbf, 4 * 512 * 512);
  k_f32_to_bf16<<<1024, 256, 0, stream>>>(rgb,    xbf,                         (int)(M_IMG * 512));
  k_f32_to_bf16<<<1024, 256, 0, stream>>>(depth,  xbf + M_IMG * 512,           (int)(M_IMG * 512));
  k_f32_to_bf16<<<1024, 256, 0, stream>>>(mmwave, xbf + 2 * M_IMG * 512,       (int)(M_PC * 512));
  k_f32_to_bf16<<<1024, 256, 0, stream>>>(lidar,  xbf + (2 * M_IMG + M_PC) * 512, (int)(M_PC * 512));
  k_prep<<<64, 256, 0, stream>>>(conv_b, bn_g, bn_b, bn_m, bn_v,
                                 lin_w_img, lin_b_img, lin_w_pc, lin_b_pc,
                                 pe_b1, pe_g1, pe_bb1, pe_m1, pe_v1,
                                 pe_w2, pe_b2, pe_g2, pe_bb2, pe_m2, pe_v2,
                                 scale, beta, linb, lwbf,
                                 pe_s1, pe_t1, pe_s2, pe_t2, w2bf);
  hipMemsetAsync(hbuf, 0, 4ull * 256 * 512 * 64 * 2, stream);  // zero K-padding

  dim3 g1(784, 4);
  k_branch_gemm<<<g1, 256, 0, stream>>>(xbf, wbf, scale, beta, hbuf);
  dim3 g2(4, 256, 4);
  k_token_linear<<<g2, 256, 0, stream>>>(hbuf, lwbf, linb, out);
  k_fps<<<256, 256, 0, stream>>>(lidar_pts, nxyz);
  k_posenc<<<256, 256, 0, stream>>>(nxyz, pe_w1, pe_s1, pe_t1, w2bf, pe_s2, pe_t2, out);
}
